// DenseFullAttention_53386443489641
// MI455X (gfx1250) — matrix operand order repaired and checked by reading, not compiled, not run on hardware
//
#include <hip/hip_runtime.h>
#include <hip/hip_bf16.h>

// ---------------------------------------------------------------------------
// Problem constants
// ---------------------------------------------------------------------------
#define BATCH   2
#define SEQ     2048
#define DMODEL  2048
#define NHEADS  16
#define HDIM    128
#define MROWS   (BATCH * SEQ)       // 4096

typedef __bf16 bf16_t;
typedef __attribute__((ext_vector_type(16))) __bf16 bf16x16;
typedef __attribute__((ext_vector_type(4)))  __bf16 bf16x4;
typedef __attribute__((ext_vector_type(8)))  float  f32x8;
typedef __attribute__((ext_vector_type(4)))  float  f32x4;
typedef __attribute__((ext_vector_type(4)))  unsigned int u32x4;
typedef __attribute__((ext_vector_type(8)))  int    i32x8;
typedef __attribute__((ext_vector_type(4)))  int    i32x4;

#if defined(__has_builtin)
#if __has_builtin(__builtin_amdgcn_tensor_load_to_lds)
#define HAVE_TDM 1
#else
#define HAVE_TDM 0
#endif
#else
#define HAVE_TDM 0
#endif

// low 32 bits of a generic pointer to LDS == LDS byte offset (flat aperture rule)
__device__ inline unsigned lds_off_u32(const void* p)
{
    return (unsigned)(unsigned long long)p;
}

// async 16B/lane copy global -> LDS (GLOBAL_LOAD_ASYNC_TO_LDS_B128, GVS mode)
__device__ inline void async_copy_b128(unsigned lds_byte, unsigned gbl_byte_off,
                                       const void* sbase)
{
    asm volatile("global_load_async_to_lds_b128 %0, %1, %2"
                 :: "v"(lds_byte), "v"(gbl_byte_off), "s"(sbase)
                 : "memory");
}
// wait until at most N async ops outstanding (in-order completion for loads)
template <int N>
__device__ inline void wait_async()
{
    asm volatile("s_wait_asynccnt %0" :: "i"(N) : "memory");
}

#if HAVE_TDM
// TDM: 2D tile load, bf16 elements (data_size = 2B).
// tensor [dim1 rows x dim0 cols], row stride stride0 (elems); tile tile1 x tile0.
__device__ inline void tdm_load_2d(unsigned lds_byte, const void* gaddr,
                                   unsigned dim0, unsigned dim1,
                                   unsigned stride0, unsigned tile0, unsigned tile1)
{
    unsigned long long ga = (unsigned long long)gaddr;
    u32x4 g0;
    g0[0] = 1u;                                         // count=1, user descriptor
    g0[1] = lds_byte;                                   // lds_addr
    g0[2] = (unsigned)ga;                               // global_addr[31:0]
    g0[3] = (unsigned)((ga >> 32) & 0x1FFFFFFull) | (2u << 30);  // addr[56:32] | type=2
    i32x8 g1;
    g1[0] = (int)(1u << 16);                            // data_size=1 (2 bytes)
    g1[1] = (int)((dim0 & 0xFFFFu) << 16);              // tensor_dim0[15:0] -> bits 63:48
    g1[2] = (int)(((dim0 >> 16) & 0xFFFFu) | ((dim1 & 0xFFFFu) << 16));
    g1[3] = (int)(((dim1 >> 16) & 0xFFFFu) | ((tile0 & 0xFFFFu) << 16));
    g1[4] = (int)(tile1 & 0xFFFFu);                     // tile_dim1 (tile_dim2 = 0)
    g1[5] = (int)stride0;                               // tensor_dim0_stride[31:0]
    g1[6] = 0;
    g1[7] = 0;
    i32x4 z4 = {0, 0, 0, 0};
    i32x8 z8 = {0, 0, 0, 0, 0, 0, 0, 0};
    __builtin_amdgcn_tensor_load_to_lds(g0, g1, z4, z4, z8, 0);
}
#endif

// ---------------------------------------------------------------------------
// f32 -> bf16 conversion (vector 4-wide)
// ---------------------------------------------------------------------------
__global__ __launch_bounds__(256)
void cvt_f32_bf16(const float* __restrict__ src, bf16_t* __restrict__ dst, int n4)
{
    int i = blockIdx.x * blockDim.x + threadIdx.x;
    if (i < n4) {
        f32x4 v = ((const f32x4*)src)[i];
        ((bf16x4*)dst)[i] = __builtin_convertvector(v, bf16x4);
    }
}

// ---------------------------------------------------------------------------
// f32 [K][N] -> bf16 transposed [N][K]  (one-time weight prep; makes all GEMM
// B-fragments contiguous so tile staging can be pure async_to_lds_b128)
// ---------------------------------------------------------------------------
__global__ __launch_bounds__(256)
void cvt_transpose_bf16(const float* __restrict__ src, bf16_t* __restrict__ dst,
                        int K, int N)
{
    __shared__ float t[32][33];
    int k0 = blockIdx.y * 32;
    int n0 = blockIdx.x * 32;
    int tx = threadIdx.x & 31;
    int ty = threadIdx.x >> 5;                // 0..7
    #pragma unroll
    for (int i = 0; i < 4; ++i) {
        int kr = ty + i * 8;
        t[kr][tx] = src[(size_t)(k0 + kr) * N + n0 + tx];
    }
    __syncthreads();
    #pragma unroll
    for (int i = 0; i < 4; ++i) {
        int nr = ty + i * 8;
        dst[(size_t)(n0 + nr) * K + k0 + tx] = (bf16_t)t[tx][nr];
    }
}

// ---------------------------------------------------------------------------
// bf16 GEMM: C[M,N] = A[M,K] * Bt[N,K]^T, fp32 accumulate, wmma 16x16x32 bf16.
// Block tile 128x128, BK=32, 256 threads (8 waves), wave tile 32x64 (2x4 WMMA).
// Double-buffered LDS staging via GLOBAL_LOAD_ASYNC_TO_LDS_B128: tile k+1
// streams in (ASYNCcnt) while tile k is consumed by the WMMA pipe.
// MODE: 0 = f32 row-major out, 1 = bf16 row-major out,
//       2 = bf16 out written as per-head-transposed V: [B][H][hd][S]
// ---------------------------------------------------------------------------
#define BM 128
#define BN 128
#define BK 32
#define TILE_BYTES (BM * BK * 2)    // 8 KB

template <int MODE>
__global__ __launch_bounds__(256)
void gemm_bf16(const bf16_t* __restrict__ A, const bf16_t* __restrict__ Bt,
               void* __restrict__ C, int M, int N, int K)
{
    __shared__ __align__(16) bf16_t lA[2][BM * BK];    // [m][k] ping-pong
    __shared__ __align__(16) bf16_t lBT[2][BN * BK];   // [n][k] ping-pong

    const int tid   = threadIdx.x;
    const int lane  = tid & 31;
    const int wave  = tid >> 5;
    const int wrow  = wave >> 1;        // 0..3 -> 32-row band
    const int wcol  = wave & 1;         // 0..1 -> 64-col band
    const int lhalf = lane >> 4;        // 0/1
    const int l16   = lane & 15;

    const int bm0 = blockIdx.y * BM;
    const int bn0 = blockIdx.x * BN;

    const unsigned ldsA0 = lds_off_u32(&lA[0][0]);
    const unsigned ldsB0 = lds_off_u32(&lBT[0][0]);

    // issue one 128x32 tile pair (A + Bt) into buffer `buf`: 4 async ops / wave
    auto issue_tiles = [&](int k0, int buf) {
        unsigned la = ldsA0 + (unsigned)buf * TILE_BYTES;
        unsigned lb = ldsB0 + (unsigned)buf * TILE_BYTES;
        #pragma unroll
        for (int c = 0; c < 2; ++c) {
            int ch = tid + c * 256;           // 512 x 16B chunks per tile
            int r  = ch >> 2;                 // 0..127
            int kk = (ch & 3) * 8;            // 0,8,16,24
            unsigned lofs = (unsigned)(r * BK + kk) * 2;
            async_copy_b128(la + lofs,
                            (unsigned)(((size_t)(bm0 + r) * K + k0 + kk) * 2), A);
            async_copy_b128(lb + lofs,
                            (unsigned)(((size_t)(bn0 + r) * K + k0 + kk) * 2), Bt);
        }
    };

    f32x8 acc[2][4] = {};

    issue_tiles(0, 0);
    const int nk = K / BK;
    for (int kt = 0; kt < nk; ++kt) {
        const int cur = kt & 1;
        if (kt + 1 < nk) {
            issue_tiles((kt + 1) * BK, 1 - cur);   // stream next tile behind compute
            // L2 warm for tile k+2 (global_prefetch_b8)
            if (kt + 2 < nk) {
                int ch = tid * 2;
                __builtin_prefetch(A  + (size_t)(bm0 + (ch >> 2)) * K + (kt + 2) * BK + (ch & 3) * 8, 0, 0);
                __builtin_prefetch(Bt + (size_t)(bn0 + (ch >> 2)) * K + (kt + 2) * BK + (ch & 3) * 8, 0, 0);
            }
            wait_async<4>();                       // tile k done; k+1 still in flight
        } else {
            wait_async<0>();
        }
        __syncthreads();

        // ---- fragments (ISA 16-bit A / B wave32 layouts)
        const bf16_t* pA = &lA[cur][0];
        const bf16_t* pB = &lBT[cur][0];
        bf16x16 afrag[2], bfrag[4];
        #pragma unroll
        for (int mi = 0; mi < 2; ++mi) {
            int row = wrow * 32 + mi * 16 + l16;
            union { bf16x16 v; u32x4 q[2]; } u;
            u.q[0] = *(const u32x4*)(pA + row * BK + lhalf * 8);       // K {0..7}/{8..15}
            u.q[1] = *(const u32x4*)(pA + row * BK + lhalf * 8 + 16);  // K {16..23}/{24..31}
            afrag[mi] = u.v;
        }
        #pragma unroll
        for (int ni = 0; ni < 4; ++ni) {
            int coln = wcol * 64 + ni * 16 + l16;
            union { bf16x16 v; u32x4 q[2]; } u;
            u.q[0] = *(const u32x4*)(pB + coln * BK + lhalf * 8);       // K {0..7}/{8..15}
            u.q[1] = *(const u32x4*)(pB + coln * BK + lhalf * 8 + 16);
            bfrag[ni] = u.v;
        }
        #pragma unroll
        for (int mi = 0; mi < 2; ++mi)
            #pragma unroll
            for (int ni = 0; ni < 4; ++ni)
                acc[mi][ni] = __builtin_amdgcn_wmma_f32_16x16x32_bf16(
                    false, afrag[mi], false, bfrag[ni],
                    (short)0, acc[mi][ni], false, false);
        __syncthreads();   // protects buffer (1-cur) before next issue overwrites it
    }

    // ---- epilogue: C layout VGPR r -> row (r + lhalf*8), lanes = N
    #pragma unroll
    for (int mi = 0; mi < 2; ++mi) {
        #pragma unroll
        for (int ni = 0; ni < 4; ++ni) {
            int n = bn0 + wcol * 64 + ni * 16 + l16;
            #pragma unroll
            for (int r = 0; r < 8; ++r) {
                int m = bm0 + wrow * 32 + mi * 16 + lhalf * 8 + r;
                if constexpr (MODE == 0) {
                    ((float*)C)[(size_t)m * N + n] = acc[mi][ni][r];
                } else if constexpr (MODE == 1) {
                    ((bf16_t*)C)[(size_t)m * N + n] = (bf16_t)acc[mi][ni][r];
                } else {
                    // V transposed per head: [B][H][hd][S]
                    int b = m >> 11, s = m & (SEQ - 1);
                    int h = n >> 7, d = n & (HDIM - 1);
                    ((bf16_t*)C)[(((size_t)(b * NHEADS + h)) * HDIM + d) * SEQ + s] =
                        (bf16_t)acc[mi][ni][r];
                }
            }
        }
    }
}

// ---------------------------------------------------------------------------
// RoPE applied in-place to bf16 Q and K, layout [B*S, H*HD]
// ---------------------------------------------------------------------------
__global__ __launch_bounds__(256)
void rope_kernel(bf16_t* __restrict__ q, bf16_t* __restrict__ k,
                 const float* __restrict__ fcos, const float* __restrict__ fsin)
{
    int t = blockIdx.x * blockDim.x + threadIdx.x;   // B*S*H*64 threads
    int i  = t & 63;
    int h  = (t >> 6) & (NHEADS - 1);
    int bs = t >> 10;
    int s  = bs & (SEQ - 1);
    size_t base = (size_t)bs * DMODEL + h * HDIM;
    float c  = fcos[s * 64 + i];
    float sn = fsin[s * 64 + i];

    float q0 = (float)q[base + i], q1 = (float)q[base + i + 64];
    q[base + i]      = (bf16_t)(q0 * c - q1 * sn);
    q[base + i + 64] = (bf16_t)(q1 * c + q0 * sn);

    float k0 = (float)k[base + i], k1 = (float)k[base + i + 64];
    k[base + i]      = (bf16_t)(k0 * c - k1 * sn);
    k[base + i + 64] = (bf16_t)(k1 * c + k0 * sn);
}

// ---------------------------------------------------------------------------
// Flash-style causal attention. Block = 8 waves = 128 queries of one (b,h).
// Double-buffered KV staging: K tile via TDM (TENSORcnt), V^T tile via
// async_to_lds (ASYNCcnt); next tile streams while current is computed.
// QK^T and P*V on v_wmma_f32_16x16x32_bf16.
// ---------------------------------------------------------------------------
__device__ inline float red_max16(float v)
{
    #pragma unroll
    for (int o = 8; o >= 1; o >>= 1) v = fmaxf(v, __shfl_xor(v, o, 16));
    return v;
}
__device__ inline float red_sum16(float v)
{
    #pragma unroll
    for (int o = 8; o >= 1; o >>= 1) v += __shfl_xor(v, o, 16);
    return v;
}

#define KV_TILE_BYTES (32 * HDIM * 2)   // 8 KB

__global__ __launch_bounds__(256)
void attn_kernel(const bf16_t* __restrict__ Q, const bf16_t* __restrict__ Kc,
                 const bf16_t* __restrict__ Vt, bf16_t* __restrict__ O)
{
    __shared__ __align__(16) bf16_t lK[2][32 * HDIM];      // [key][hd]  2 x 8 KB
    __shared__ __align__(16) bf16_t lVT[2][HDIM * 32];     // [hd][key]  2 x 8 KB
    __shared__ __align__(16) bf16_t lP[8][16 * 32];        // per-wave P 8 KB

    const int tid   = threadIdx.x;
    const int lane  = tid & 31;
    const int wave  = tid >> 5;
    const int lhalf = lane >> 4;
    const int l16   = lane & 15;

    const int Q0 = blockIdx.x * 128;
    const int h  = blockIdx.y;
    const int b  = blockIdx.z;
    const int q0 = Q0 + wave * 16;

    const float scale = 0.08838834764831845f;   // 1/sqrt(128)

    const bf16_t* Kb  = Kc + (size_t)b * SEQ * DMODEL + h * HDIM;          // row stride DMODEL
    const bf16_t* Vtb = Vt + ((size_t)(b * NHEADS + h)) * HDIM * SEQ;      // [hd][S]
    const unsigned ldsK0 = lds_off_u32(&lK[0][0]);
    const unsigned ldsV0 = lds_off_u32(&lVT[0][0]);

    // issue one 32-key KV tile into buffer `buf`
    auto issue_kv = [&](int k0, int buf) {
#if HAVE_TDM
        if (wave == 0) {
            tdm_load_2d(ldsK0 + (unsigned)buf * KV_TILE_BYTES,
                        Kb + (size_t)k0 * DMODEL,
                        /*dim0*/HDIM, /*dim1*/(unsigned)(SEQ - k0),
                        /*stride0*/DMODEL, /*tile0*/HDIM, /*tile1*/32);
        }
#else
        #pragma unroll
        for (int c = 0; c < 2; ++c) {
            int ch   = tid + c * 256;
            int key  = ch >> 4;
            int col8 = (ch & 15) * 8;
            async_copy_b128(ldsK0 + (unsigned)buf * KV_TILE_BYTES
                                  + (unsigned)(key * HDIM + col8) * 2,
                            (unsigned)(((size_t)(k0 + key) * DMODEL + col8) * 2), Kb);
        }
#endif
        #pragma unroll
        for (int c = 0; c < 2; ++c) {
            int ch = tid + c * 256;
            int d  = ch >> 2;                 // 0..127
            int ko = (ch & 3) * 8;            // 0,8,16,24
            async_copy_b128(ldsV0 + (unsigned)buf * KV_TILE_BYTES
                                  + (unsigned)(d * 32 + ko) * 2,
                            (unsigned)(((size_t)d * SEQ + k0 + ko) * 2), Vtb);
        }
    };

    // load Q fragments once: 4 x (16x32) A-frags covering hd=128
    bf16x16 qf[4];
    {
        size_t base = ((size_t)(b * SEQ + q0 + l16)) * DMODEL + h * HDIM;
        #pragma unroll
        for (int j = 0; j < 4; ++j) {
            union { bf16x16 v; u32x4 q[2]; } u;
            u.q[0] = *(const u32x4*)(Q + base + j * 32 + lhalf * 8);
            u.q[1] = *(const u32x4*)(Q + base + j * 32 + lhalf * 8 + 16);
            qf[j] = u.v;
        }
    }

    f32x8 outacc[8] = {};
    float mrow[8], lrow[8];
    #pragma unroll
    for (int r = 0; r < 8; ++r) { mrow[r] = -1e30f; lrow[r] = 0.0f; }

    const int nkt = Q0 / 32 + 4;
    issue_kv(0, 0);
    for (int kt = 0; kt < nkt; ++kt) {
        const int k0  = kt * 32;
        const int cur = kt & 1;

        if (kt + 1 < nkt) {
            issue_kv((kt + 1) * 32, 1 - cur);     // stream next KV tile
#if HAVE_TDM
            wait_async<2>();
            if (wave == 0) __builtin_amdgcn_s_wait_tensorcnt((short)1);
#else
            wait_async<4>();
#endif
        } else {
            wait_async<0>();
#if HAVE_TDM
            if (wave == 0) __builtin_amdgcn_s_wait_tensorcnt((short)0);
#endif
        }
        __syncthreads();

        if (k0 <= q0 + 15) {
            const bf16_t* pK = &lK[cur][0];
            const bf16_t* pV = &lVT[cur][0];

            // ---- S = Q K^T  (two 16x16 tiles: keys k0..15 and k0+16..31)
            f32x8 s0 = {}, s1 = {};
            #pragma unroll
            for (int j = 0; j < 4; ++j) {
                union { bf16x16 v; u32x4 q[2]; } u;
                u.q[0] = *(const u32x4*)(pK + l16 * HDIM + j * 32 + lhalf * 8);
                u.q[1] = *(const u32x4*)(pK + l16 * HDIM + j * 32 + lhalf * 8 + 16);
                s0 = __builtin_amdgcn_wmma_f32_16x16x32_bf16(false, qf[j], false, u.v,
                                                             (short)0, s0, false, false);
                u.q[0] = *(const u32x4*)(pK + (16 + l16) * HDIM + j * 32 + lhalf * 8);
                u.q[1] = *(const u32x4*)(pK + (16 + l16) * HDIM + j * 32 + lhalf * 8 + 16);
                s1 = __builtin_amdgcn_wmma_f32_16x16x32_bf16(false, qf[j], false, u.v,
                                                             (short)0, s1, false, false);
            }

            // ---- online softmax (rows r / r+8 per C layout, lanes = key)
            float alpha[8];
            #pragma unroll
            for (int r = 0; r < 8; ++r) {
                int m    = r + lhalf * 8;
                int qrow = q0 + m;
                float v0 = s0[r] * scale;
                if (k0 + l16 > qrow)      v0 = -1e30f;
                float v1 = s1[r] * scale;
                if (k0 + 16 + l16 > qrow) v1 = -1e30f;
                float tmax = red_max16(fmaxf(v0, v1));
                float mnew = fmaxf(mrow[r], tmax);
                float al   = __expf(mrow[r] - mnew);
                float p0   = __expf(v0 - mnew);
                float p1   = __expf(v1 - mnew);
                lrow[r] = lrow[r] * al + red_sum16(p0 + p1);
                mrow[r] = mnew;
                alpha[r] = al;
                lP[wave][m * 32 + l16]      = (bf16_t)p0;   // same-wave DS: in-order
                lP[wave][m * 32 + 16 + l16] = (bf16_t)p1;
            }

            // rescale existing output accumulators
            #pragma unroll
            for (int c = 0; c < 8; ++c)
                #pragma unroll
                for (int r = 0; r < 8; ++r)
                    outacc[c][r] *= alpha[r];

            // ---- P as A-frag (16x32), V^T as B-frags, out += P*V
            union { bf16x16 v; u32x4 q[2]; } up;
            up.q[0] = *(const u32x4*)(lP[wave] + l16 * 32 + lhalf * 8);
            up.q[1] = *(const u32x4*)(lP[wave] + l16 * 32 + lhalf * 8 + 16);
            bf16x16 pf = up.v;
            #pragma unroll
            for (int c = 0; c < 8; ++c) {
                union { bf16x16 v; u32x4 q[2]; } uv;
                uv.q[0] = *(const u32x4*)(pV + (c * 16 + l16) * 32 + lhalf * 8);
                uv.q[1] = *(const u32x4*)(pV + (c * 16 + l16) * 32 + lhalf * 8 + 16);
                outacc[c] = __builtin_amdgcn_wmma_f32_16x16x32_bf16(false, pf, false, uv.v,
                                                                    (short)0, outacc[c], false, false);
            }
        }
        __syncthreads();   // protects buffer (1-cur) before next issue overwrites it
    }

    // ---- normalize and store attention output (bf16, [B*S, H*HD])
    #pragma unroll
    for (int r = 0; r < 8; ++r) {
        int m = r + lhalf * 8;
        float inv = 1.0f / lrow[r];
        size_t base = ((size_t)(b * SEQ + q0 + m)) * DMODEL + h * HDIM;
        #pragma unroll
        for (int c = 0; c < 8; ++c)
            O[base + c * 16 + l16] = (bf16_t)(outacc[c][r] * inv);
    }
}

// ---------------------------------------------------------------------------
// Launcher
// ---------------------------------------------------------------------------
extern "C" void kernel_launch(void* const* d_in, const int* in_sizes, int n_in,
                              void* d_out, int out_size, void* d_ws, size_t ws_size,
                              hipStream_t stream)
{
    const float* x    = (const float*)d_in[0];
    const float* fcos = (const float*)d_in[1];
    const float* fsin = (const float*)d_in[2];
    const float* wq   = (const float*)d_in[3];
    const float* wk   = (const float*)d_in[4];
    const float* wv   = (const float*)d_in[5];
    const float* wo   = (const float*)d_in[6];
    float* out = (float*)d_out;

    char* ws = (char*)d_ws;
    const size_t MB = 1ull << 20;
    bf16_t* xb  = (bf16_t*)(ws +  0 * MB);   // 16 MB  x in bf16
    bf16_t* wqt = (bf16_t*)(ws + 16 * MB);   //  8 MB  wq^T bf16 [N][K]
    bf16_t* wkt = (bf16_t*)(ws + 24 * MB);   //  8 MB
    bf16_t* wvt = (bf16_t*)(ws + 32 * MB);   //  8 MB
    bf16_t* wot = (bf16_t*)(ws + 40 * MB);   //  8 MB
    bf16_t* qb  = (bf16_t*)(ws + 48 * MB);   // 16 MB  Q [B*S, H*HD]
    bf16_t* kb  = (bf16_t*)(ws + 64 * MB);   // 16 MB  K [B*S, H*HD]
    bf16_t* vt  = (bf16_t*)(ws + 80 * MB);   // 16 MB  V^T [B][H][hd][S]
    bf16_t* ab  = (bf16_t*)(ws + 96 * MB);   // 16 MB  attention out

    // conversions (x plain; weights transposed for async-friendly GEMM staging)
    int nx4 = BATCH * SEQ * DMODEL / 4;
    cvt_f32_bf16<<<nx4 / 256, 256, 0, stream>>>(x, xb, nx4);
    dim3 gt(DMODEL / 32, DMODEL / 32);        // (64, 64)
    cvt_transpose_bf16<<<gt, 256, 0, stream>>>(wq, wqt, DMODEL, DMODEL);
    cvt_transpose_bf16<<<gt, 256, 0, stream>>>(wk, wkt, DMODEL, DMODEL);
    cvt_transpose_bf16<<<gt, 256, 0, stream>>>(wv, wvt, DMODEL, DMODEL);
    cvt_transpose_bf16<<<gt, 256, 0, stream>>>(wo, wot, DMODEL, DMODEL);

    // QKV projections
    dim3 gg(DMODEL / BN, MROWS / BM);         // (16, 32)
    gemm_bf16<1><<<gg, 256, 0, stream>>>(xb, wqt, qb, MROWS, DMODEL, DMODEL);
    gemm_bf16<1><<<gg, 256, 0, stream>>>(xb, wkt, kb, MROWS, DMODEL, DMODEL);
    gemm_bf16<2><<<gg, 256, 0, stream>>>(xb, wvt, vt, MROWS, DMODEL, DMODEL);

    // RoPE on Q and K
    int nrope = BATCH * SEQ * NHEADS * 64;
    rope_kernel<<<nrope / 256, 256, 0, stream>>>(qb, kb, fcos, fsin);

    // causal flash attention
    attn_kernel<<<dim3(SEQ / 128, NHEADS, BATCH), 256, 0, stream>>>(qb, kb, vt, ab);

    // output projection (fp32 out)
    gemm_bf16<0><<<gg, 256, 0, stream>>>(ab, wot, out, MROWS, DMODEL, DMODEL);
}
